// LVCBlock_43087111913877
// MI455X (gfx1250) — matrix operand order repaired and checked by reading, not compiled, not run on hardware
//
#include <hip/hip_runtime.h>
#include <math.h>

// ---------------------------------------------------------------------------
// Problem constants (fixed by the reference)
// ---------------------------------------------------------------------------
#define BB_   16
#define CC_   512
#define C4_   128
#define HH_   64
#define WW_   64
#define HW_   4096           // H*W
#define NP_   65536          // B*H*W (total pixels)
#define KK_   64             // num codes
#define EPS_CB 1e-6f
#define EPS_BN 1e-5f

typedef __attribute__((ext_vector_type(16))) __bf16 v16bf;
typedef __attribute__((ext_vector_type(8)))  float  v8f;

// fp32 <-> bf16 helpers (RNE truncation)
__device__ __forceinline__ __bf16 f2bf(float f) {
  unsigned u = __builtin_bit_cast(unsigned, f);
  unsigned r = u + 0x7FFFu + ((u >> 16) & 1u);
  unsigned short h = (unsigned short)(r >> 16);
  return __builtin_bit_cast(__bf16, h);
}
__device__ __forceinline__ float bf2f(__bf16 b) {
  unsigned short s = __builtin_bit_cast(unsigned short, b);
  unsigned u = ((unsigned)s) << 16;
  return __builtin_bit_cast(float, u);
}

// ---------------------------------------------------------------------------
// Async global->LDS staging (gfx1250 ASYNCcnt path), 16-byte chunks.
// Builtin signature (from the toolchain diagnostic): first param is an
// int4-vector pointer in the global (AS1, printed "__device__") address
// space; second is the LDS (AS3) destination.  Falls back to a plain b128
// copy through VGPRs if the builtin is absent.
// ---------------------------------------------------------------------------
#if defined(__has_builtin)
#  if __has_builtin(__builtin_amdgcn_global_load_async_to_lds_b128)
#    define USE_ASYNC 1
#  endif
#endif
#ifndef USE_ASYNC
#  define USE_ASYNC 0
#endif

typedef int v4i_ __attribute__((vector_size(16)));

__device__ __forceinline__ void copy16(const __bf16* gsrc, __bf16* ldst) {
#if USE_ASYNC
  typedef __attribute__((address_space(1))) v4i_* gp_t;
  typedef __attribute__((address_space(3))) v4i_* lp_t;
  __builtin_amdgcn_global_load_async_to_lds_b128((gp_t)(gsrc), (lp_t)(ldst), 0, 0);
#else
  *(v4i_*)ldst = *(const v4i_*)gsrc;
#endif
}
__device__ __forceinline__ void async_wait0() {
#if USE_ASYNC
#  if __has_builtin(__builtin_amdgcn_s_wait_asynccnt)
  __builtin_amdgcn_s_wait_asynccnt(0);
#  else
  asm volatile("s_wait_asynccnt 0x0" ::: "memory");
#  endif
#endif
}

// ---------------------------------------------------------------------------
// WMMA fragment loaders (ISA 7.12.2, 16-bit operand layouts, wave32).
// LDS tiles are [64][32] bf16 (64-B rows, every 16-B chunk aligned).
// ---------------------------------------------------------------------------
__device__ __forceinline__ v16bf load_a_frag(const __bf16* s, int mt, int lane) {
  int row = mt * 16 + (lane & 15);
  int hi  = (lane >> 4) << 3;           // 0 or 8
  const __bf16* p = s + row * 32;
  v16bf a;
#pragma unroll
  for (int i = 0; i < 16; ++i) {
    int k = ((i < 8) ? i : (i + 8)) + hi;
    a[i] = p[k];
  }
  return a;
}
__device__ __forceinline__ v16bf load_b_frag(const __bf16* s, int nt, int lane) {
  int col = nt * 16 + (lane & 15);
  int kb  = (lane >> 4) << 3;           // 0 or 8
  const __bf16* p = s + col * 32 + kb;
  v16bf b;
#pragma unroll
  for (int i = 0; i < 16; ++i) b[i] = p[(i < 8) ? i : (i + 8)];
  return b;
}

// ---------------------------------------------------------------------------
// Generic WMMA GEMM: acc[m][n] = sum_k A[m0+m][k0+k] * Wt[n0+n][k0+k]
// A, Wt bf16 row-major (K-contiguous).  Epilogue modes:
//   0: Ob  = bf16(relu(acc*s + t))                       (BN+ReLU)
//   1: Out = acc*s + t                                   (BN, fp32)
//   2: Out += relu(acc*s + t); Ob = bf16(Out)            (BN+ReLU accumulate)
//   3: Out = sc[n]*(xsq[m] - 2*acc + csq[n])             (scaled L2 distance)
//   4: Out = sigmoid(acc + bb[n])                        (fc gate)
//   5: v = relu(acc*s + t); Ob = bf16(v) pixel-major; Ot = bf16(v) channel-major
// NOTE: A must be allocated with rows padded to a multiple of 64 (fc case).
// ---------------------------------------------------------------------------
__global__ void __launch_bounds__(256)
gemm_bf16_kernel(const __bf16* __restrict__ A, int lda,
                 const __bf16* __restrict__ Wt, int ldw,
                 float* __restrict__ Out, __bf16* __restrict__ Ob,
                 __bf16* __restrict__ Ot, int ldo,
                 int M, int Kdim, int mode, float eps,
                 const float* __restrict__ g,  const float* __restrict__ bb,
                 const float* __restrict__ mm, const float* __restrict__ vv,
                 const float* __restrict__ xsq, const float* __restrict__ csq,
                 const float* __restrict__ sc) {
  __shared__ __bf16 As[64 * 32];
  __shared__ __bf16 Bs[64 * 32];
  const int tid  = threadIdx.x;
  const int lane = tid & 31;
  const int wv   = tid >> 5;
  const int mt   = wv >> 1;
  const int ntA  = (wv & 1) * 2;
  const int m0   = blockIdx.x * 64;
  const int n0   = blockIdx.y * 64;
  const int r    = tid >> 2;            // staged row (0..63)
  const int cb   = (tid & 3) * 8;       // staged chunk (8 bf16 = 16 B)

  v8f acc0 = {}, acc1 = {};

  for (int k0 = 0; k0 < Kdim; k0 += 32) {
    copy16(A  + (size_t)(m0 + r) * lda + k0 + cb, &As[r * 32 + cb]);
    copy16(Wt + (size_t)(n0 + r) * ldw + k0 + cb, &Bs[r * 32 + cb]);
    async_wait0();
    __syncthreads();
    v16bf af = load_a_frag(As, mt, lane);
    v16bf b0 = load_b_frag(Bs, ntA, lane);
    v16bf b1 = load_b_frag(Bs, ntA + 1, lane);
    acc0 = __builtin_amdgcn_wmma_f32_16x16x32_bf16(false, af, false, b0,
                                                   (short)0, acc0, false, false);
    acc1 = __builtin_amdgcn_wmma_f32_16x16x32_bf16(false, af, false, b1,
                                                   (short)0, acc1, false, false);
    __syncthreads();
  }

  const int mBase = m0 + mt * 16 + ((lane >> 4) << 3);

  auto store_tile = [&](v8f acc, int nt) {
    int gn = n0 + nt * 16 + (lane & 15);
    float s = 1.f, t = 0.f;
    if (mode <= 2 || mode == 5) {
      float sv = g[gn] * rsqrtf(vv[gn] + eps);
      s = sv; t = bb[gn] - mm[gn] * sv;
    }
    float scn = 0.f, cqn = 0.f, bn4 = 0.f;
    if (mode == 3) { scn = sc[gn]; cqn = csq[gn]; }
    if (mode == 4) { bn4 = bb[gn]; }
#pragma unroll
    for (int rr = 0; rr < 8; ++rr) {
      int gm = mBase + rr;
      if (gm >= M) continue;
      size_t oi = (size_t)gm * ldo + gn;
      float a = acc[rr];
      if (mode == 0) {
        Ob[oi] = f2bf(fmaxf(0.f, a * s + t));
      } else if (mode == 1) {
        Out[oi] = a * s + t;
      } else if (mode == 2) {
        float nv = Out[oi] + fmaxf(0.f, a * s + t);
        Out[oi] = nv; Ob[oi] = f2bf(nv);
      } else if (mode == 3) {
        Out[oi] = scn * (xsq[gm] - 2.f * a + cqn);
      } else if (mode == 4) {
        Out[oi] = 1.f / (1.f + __expf(-(a + bn4)));
      } else {
        float v = fmaxf(0.f, a * s + t);
        __bf16 bv = f2bf(v);
        Ob[oi] = bv;
        Ot[((size_t)(gm >> 12) * CC_ + gn) * HW_ + (gm & 4095)] = bv;
      }
    }
  };
  store_tile(acc0, ntA);
  store_tile(acc1, ntA + 1);
}

// ---------------------------------------------------------------------------
// 3x3 conv (pad=1) implicit GEMM. Tile = one image row (64 px) x 64 outch;
// 9 taps x 4 slices of 32 input channels.  In/Out bf16 NHWC; W2b pre-
// reordered to [tap][outch][inch] bf16 so tiles are K-contiguous.
// ---------------------------------------------------------------------------
__global__ void __launch_bounds__(256)
conv3x3_bn_relu_kernel(const __bf16* __restrict__ In,
                       const __bf16* __restrict__ W2b,
                       __bf16* __restrict__ Outb,
                       const float* __restrict__ g,  const float* __restrict__ bb,
                       const float* __restrict__ mm, const float* __restrict__ vv,
                       float eps) {
  __shared__ __bf16 As[64 * 32];
  __shared__ __bf16 Bs[64 * 32];
  const int tid  = threadIdx.x;
  const int lane = tid & 31;
  const int wv   = tid >> 5;
  const int mt   = wv >> 1;
  const int ntA  = (wv & 1) * 2;
  const int b    = blockIdx.x >> 6;
  const int h    = blockIdx.x & 63;
  const int n0   = blockIdx.y * 64;
  const int r    = tid >> 2;
  const int cb   = (tid & 3) * 8;

  v8f acc0 = {}, acc1 = {};
  const v4i_ z4 = {0, 0, 0, 0};

  for (int tap = 0; tap < 9; ++tap) {
    int dy = tap / 3 - 1, dx = tap % 3 - 1;
    int hh = h + dy;
    bool rowok = (hh >= 0) && (hh < HH_);
    for (int kc = 0; kc < C4_; kc += 32) {
      int ww = r + dx;
      if (rowok && (unsigned)ww < (unsigned)WW_)
        copy16(In + (((size_t)b * HH_ + hh) * WW_ + ww) * C4_ + kc + cb,
               &As[r * 32 + cb]);
      else
        *(v4i_*)&As[r * 32 + cb] = z4;
      copy16(W2b + ((size_t)tap * C4_ + n0 + r) * C4_ + kc + cb,
             &Bs[r * 32 + cb]);
      async_wait0();
      __syncthreads();
      v16bf af = load_a_frag(As, mt, lane);
      v16bf b0 = load_b_frag(Bs, ntA, lane);
      v16bf b1 = load_b_frag(Bs, ntA + 1, lane);
      acc0 = __builtin_amdgcn_wmma_f32_16x16x32_bf16(false, af, false, b0,
                                                     (short)0, acc0, false, false);
      acc1 = __builtin_amdgcn_wmma_f32_16x16x32_bf16(false, af, false, b1,
                                                     (short)0, acc1, false, false);
      __syncthreads();
    }
  }

  const int wBase = mt * 16 + ((lane >> 4) << 3);
  auto store_tile = [&](v8f acc, int nt) {
    int gn = n0 + nt * 16 + (lane & 15);
    float sv = g[gn] * rsqrtf(vv[gn] + eps);
    float tv = bb[gn] - mm[gn] * sv;
#pragma unroll
    for (int rr = 0; rr < 8; ++rr) {
      size_t pix = ((size_t)b * HH_ + h) * WW_ + (wBase + rr);
      Outb[pix * C4_ + gn] = f2bf(fmaxf(0.f, acc[rr] * sv + tv));
    }
  };
  store_tile(acc0, ntA);
  store_tile(acc1, ntA + 1);
}

// ---------------------------------------------------------------------------
// Encoding aggregation: enc[b,k,c] = sum_n A[b,n,k]*y[b,n,c] - asum[b,k]*cw[k,c]
// + BN1d over codes + ReLU -> encR (B*64, 512) fp32.
// AbT: (B,64,4096) bf16 code-major; Yt: (B,512,4096) bf16 channel-major.
// ---------------------------------------------------------------------------
__global__ void __launch_bounds__(256)
enc_gemm_kernel(const __bf16* __restrict__ AbT,
                const __bf16* __restrict__ Yt,
                const float* __restrict__ asum,
                const float* __restrict__ cw,
                const float* __restrict__ ge, const float* __restrict__ be_,
                const float* __restrict__ me, const float* __restrict__ ve,
                float eps, float* __restrict__ encR) {
  __shared__ __bf16 As[64 * 32];
  __shared__ __bf16 Bs[64 * 32];
  const int tid  = threadIdx.x;
  const int lane = tid & 31;
  const int wv   = tid >> 5;
  const int mt   = wv >> 1;
  const int ntA  = (wv & 1) * 2;
  const int b    = blockIdx.x;
  const int n0   = blockIdx.y * 64;
  const int r    = tid >> 2;
  const int cb   = (tid & 3) * 8;

  v8f acc0 = {}, acc1 = {};

  for (int k0 = 0; k0 < HW_; k0 += 32) {
    copy16(AbT + ((size_t)b * KK_ + r) * HW_ + k0 + cb, &As[r * 32 + cb]);
    copy16(Yt  + ((size_t)b * CC_ + n0 + r) * HW_ + k0 + cb, &Bs[r * 32 + cb]);
    async_wait0();
    __syncthreads();
    v16bf af = load_a_frag(As, mt, lane);
    v16bf b0 = load_b_frag(Bs, ntA, lane);
    v16bf b1 = load_b_frag(Bs, ntA + 1, lane);
    acc0 = __builtin_amdgcn_wmma_f32_16x16x32_bf16(false, af, false, b0,
                                                   (short)0, acc0, false, false);
    acc1 = __builtin_amdgcn_wmma_f32_16x16x32_bf16(false, af, false, b1,
                                                   (short)0, acc1, false, false);
    __syncthreads();
  }

  const int laneHi = (lane >> 4) << 3;
  auto store_tile = [&](v8f acc, int nt) {
    int gch = n0 + nt * 16 + (lane & 15);
#pragma unroll
    for (int rr = 0; rr < 8; ++rr) {
      int code = mt * 16 + laneHi + rr;
      float se = ge[code] * rsqrtf(ve[code] + eps);
      float te = be_[code] - me[code] * se;
      float ev = acc[rr] - asum[b * KK_ + code] * cw[(size_t)code * CC_ + gch];
      encR[((size_t)b * KK_ + code) * CC_ + gch] = fmaxf(0.f, ev * se + te);
    }
  };
  store_tile(acc0, ntA);
  store_tile(acc1, ntA + 1);
}

// ---------------------------------------------------------------------------
// Small helper kernels
// ---------------------------------------------------------------------------
__global__ void cvt_bf16_kernel(const float* __restrict__ src,
                                __bf16* __restrict__ dst, int n) {
  int i = blockIdx.x * blockDim.x + threadIdx.x;
  if (i < n) dst[i] = f2bf(src[i]);
}

// w2 (128,128,3,3) fp32 -> (tap, outch, inch) bf16
__global__ void w2_reorder_kernel(const float* __restrict__ src,
                                  __bf16* __restrict__ dst) {
  int i = blockIdx.x * blockDim.x + threadIdx.x;
  if (i >= C4_ * C4_ * 9) return;
  int o = i / (C4_ * 9), rem = i % (C4_ * 9);
  int ci = rem / 9, tap = rem % 9;
  dst[((size_t)tap * C4_ + o) * C4_ + ci] = f2bf(src[i]);
}

// NCHW fp32 -> NHWC bf16 through LDS (32x32 tiles)
__global__ void __launch_bounds__(256)
transpose_fwd_kernel(const float* __restrict__ X, __bf16* __restrict__ XTb) {
  __shared__ float t[32][33];
  int b = blockIdx.x, c0 = blockIdx.y * 32, p0 = blockIdx.z * 32;
  int tx = threadIdx.x, ty = threadIdx.y;
#pragma unroll
  for (int j = 0; j < 4; ++j) {
    int c = c0 + ty + j * 8;
    t[ty + j * 8][tx] = X[((size_t)b * CC_ + c) * HW_ + p0 + tx];
  }
  __syncthreads();
#pragma unroll
  for (int j = 0; j < 4; ++j) {
    int p = p0 + ty + j * 8;
    XTb[((size_t)b * HW_ + p) * CC_ + c0 + tx] = f2bf(t[tx][ty + j * 8]);
  }
}

// per-row sum of squares over bf16 rows (one wave per row)
__global__ void __launch_bounds__(256)
rowsq_bf16_kernel(const __bf16* __restrict__ src, float* __restrict__ dst,
                  int rows, int cols) {
  int row = blockIdx.x * 8 + (threadIdx.x >> 5);
  int lane = threadIdx.x & 31;
  if (row >= rows) return;
  const __bf16* rp = src + (size_t)row * cols;
  float s = 0.f;
  for (int c = lane; c < cols; c += 32) { float v = bf2f(rp[c]); s += v * v; }
#pragma unroll
  for (int o = 16; o > 0; o >>= 1) s += __shfl_xor(s, o);
  if (lane == 0) dst[row] = s;
}

// softmax over K=64 per pixel; writes code-major bf16 A^T (B,64,4096)
__global__ void __launch_bounds__(256)
softmax64_kernel(const float* __restrict__ D, __bf16* __restrict__ AbT,
                 int rows) {
  int row = blockIdx.x * 8 + (threadIdx.x >> 5);
  int lane = threadIdx.x & 31;
  if (row >= rows) return;
  const float* rp = D + (size_t)row * KK_;
  float v0 = rp[lane], v1 = rp[lane + 32];
  float mx = fmaxf(v0, v1);
#pragma unroll
  for (int o = 16; o > 0; o >>= 1) mx = fmaxf(mx, __shfl_xor(mx, o));
  v0 = __expf(v0 - mx); v1 = __expf(v1 - mx);
  float sm = v0 + v1;
#pragma unroll
  for (int o = 16; o > 0; o >>= 1) sm += __shfl_xor(sm, o);
  float inv = 1.f / sm;
  int b = row >> 12, n = row & 4095;
  __bf16* op = AbT + ((size_t)b * KK_) * HW_ + n;
  op[(size_t)lane * HW_]        = f2bf(v0 * inv);
  op[(size_t)(lane + 32) * HW_] = f2bf(v1 * inv);
}

// asum[b,k] = sum_n A^T[b,k,n]  (contiguous row reduction, one wave per row)
__global__ void __launch_bounds__(256)
asum_kernel(const __bf16* __restrict__ AbT, float* __restrict__ asum) {
  int row = blockIdx.x * 8 + (threadIdx.x >> 5);   // B*K = 1024 rows
  int lane = threadIdx.x & 31;
  if (row >= BB_ * KK_) return;
  const __bf16* rp = AbT + (size_t)row * HW_;
  float s = 0.f;
  for (int c = lane; c < HW_; c += 32) s += bf2f(rp[c]);
#pragma unroll
  for (int o = 16; o > 0; o >>= 1) s += __shfl_xor(s, o);
  if (lane == 0) asum[row] = s;
}

// en[b,c] = mean over 64 codes of encR; stored bf16 (fc GEMM input)
__global__ void mean_codes_kernel(const float* __restrict__ encR,
                                  __bf16* __restrict__ en) {
  int i = blockIdx.x * blockDim.x + threadIdx.x;
  if (i >= BB_ * CC_) return;
  int b = i >> 9, c = i & 511;
  float s = 0.f;
#pragma unroll 8
  for (int k = 0; k < KK_; ++k) s += encR[((size_t)b * KK_ + k) * CC_ + c];
  en[i] = f2bf(s * (1.f / KK_));
}

// out[b,c,hw] = relu(x1[b,hw,c] * (1 + gamma[b,c]))  (NHWC -> NCHW via LDS)
__global__ void __launch_bounds__(256)
final_gate_kernel(const float* __restrict__ X1T, const float* __restrict__ gam,
                  float* __restrict__ Out) {
  __shared__ float t[32][33];
  int b = blockIdx.x, c0 = blockIdx.y * 32, p0 = blockIdx.z * 32;
  int tx = threadIdx.x, ty = threadIdx.y;
#pragma unroll
  for (int j = 0; j < 4; ++j) {
    int p = p0 + ty + j * 8;
    int c = c0 + tx;
    float v = X1T[((size_t)b * HW_ + p) * CC_ + c];
    float g = gam[b * CC_ + c];
    t[ty + j * 8][tx] = fmaxf(0.f, v * (1.f + g));
  }
  __syncthreads();
#pragma unroll
  for (int j = 0; j < 4; ++j) {
    int c = c0 + ty + j * 8;
    Out[((size_t)b * CC_ + c) * HW_ + p0 + tx] = t[tx][ty + j * 8];
  }
}

// ---------------------------------------------------------------------------
// Host launcher
// ---------------------------------------------------------------------------
extern "C" void kernel_launch(void* const* d_in, const int* in_sizes, int n_in,
                              void* d_out, int out_size, void* d_ws, size_t ws_size,
                              hipStream_t stream) {
  (void)in_sizes; (void)n_in; (void)out_size; (void)ws_size;
  const float* x   = (const float*)d_in[0];
  const float* w1  = (const float*)d_in[1];
  const float* g1  = (const float*)d_in[2];
  const float* b1  = (const float*)d_in[3];
  const float* m1  = (const float*)d_in[4];
  const float* v1  = (const float*)d_in[5];
  const float* w2  = (const float*)d_in[6];
  const float* g2  = (const float*)d_in[7];
  const float* b2  = (const float*)d_in[8];
  const float* m2  = (const float*)d_in[9];
  const float* v2  = (const float*)d_in[10];
  const float* w3  = (const float*)d_in[11];
  const float* g3  = (const float*)d_in[12];
  const float* b3  = (const float*)d_in[13];
  const float* m3  = (const float*)d_in[14];
  const float* v3  = (const float*)d_in[15];
  const float* wr  = (const float*)d_in[16];
  const float* gr  = (const float*)d_in[17];
  const float* br  = (const float*)d_in[18];
  const float* mr  = (const float*)d_in[19];
  const float* vr  = (const float*)d_in[20];
  const float* wl  = (const float*)d_in[21];
  const float* gl  = (const float*)d_in[22];
  const float* bl  = (const float*)d_in[23];
  const float* ml  = (const float*)d_in[24];
  const float* vl  = (const float*)d_in[25];
  const float* cw  = (const float*)d_in[26];
  const float* scl = (const float*)d_in[27];
  const float* ge  = (const float*)d_in[28];
  const float* be_ = (const float*)d_in[29];
  const float* me  = (const float*)d_in[30];
  const float* ve  = (const float*)d_in[31];
  const float* wfc = (const float*)d_in[32];
  const float* bfc = (const float*)d_in[33];
  float* out = (float*)d_out;

  // ---- workspace carve-up (256-B aligned), with region reuse ----
  char* p = (char*)d_ws;
  auto alloc = [&](size_t bytes) {
    char* r = p; p += (bytes + 255) & ~(size_t)255; return r;
  };
  float*  x1T  = (float*) alloc((size_t)NP_ * CC_ * 4);   // fp32 x1 (final)
  __bf16* xTb  = (__bf16*)alloc((size_t)NP_ * CC_ * 2);   // x NHWC; reused: yT
  __bf16* x1b  = (__bf16*)alloc((size_t)NP_ * CC_ * 2);   // bf16 x1 (wl input)
  __bf16* y    = (__bf16*)alloc((size_t)NP_ * CC_ * 2);   // LVC out, pixel-major
  __bf16* h1b  = (__bf16*)alloc((size_t)NP_ * C4_ * 2);   // reused: AbT (8 MB)
  __bf16* h2b  = (__bf16*)alloc((size_t)NP_ * C4_ * 2);   // reused: dist (16 MB)
  __bf16* w1b  = (__bf16*)alloc((size_t)C4_ * CC_ * 2);
  __bf16* w2b  = (__bf16*)alloc((size_t)9 * C4_ * C4_ * 2);
  __bf16* w3b  = (__bf16*)alloc((size_t)CC_ * C4_ * 2);
  __bf16* wrb  = (__bf16*)alloc((size_t)CC_ * CC_ * 2);
  __bf16* wlb  = (__bf16*)alloc((size_t)CC_ * CC_ * 2);
  __bf16* cwb  = (__bf16*)alloc((size_t)KK_ * CC_ * 2);
  __bf16* wfcb = (__bf16*)alloc((size_t)CC_ * CC_ * 2);
  float*  xsq  = (float*) alloc((size_t)NP_ * 4);
  float*  csq  = (float*) alloc((size_t)KK_ * 4);
  float*  asum = (float*) alloc((size_t)BB_ * KK_ * 4);
  float*  encR = (float*) alloc((size_t)BB_ * KK_ * CC_ * 4);
  __bf16* en   = (__bf16*)alloc((size_t)64 * CC_ * 2);    // padded to 64 rows
  float*  gam  = (float*) alloc((size_t)BB_ * CC_ * 4);
  __bf16* yT   = xTb;                                     // reuse (xTb dead)
  float*  dist = (float*)h2b;                             // reuse (h2b dead)
  __bf16* AbT  = h1b;                                     // reuse (h1b dead)

  dim3 tb(32, 8, 1);
  dim3 tg(BB_, CC_ / 32, HW_ / 32);

  // 0) weight conversion to bf16 (once per launch; tiny)
  cvt_bf16_kernel<<<(C4_ * CC_ + 255) / 256, 256, 0, stream>>>(w1, w1b, C4_ * CC_);
  w2_reorder_kernel<<<(9 * C4_ * C4_ + 255) / 256, 256, 0, stream>>>(w2, w2b);
  cvt_bf16_kernel<<<(CC_ * C4_ + 255) / 256, 256, 0, stream>>>(w3, w3b, CC_ * C4_);
  cvt_bf16_kernel<<<(CC_ * CC_ + 255) / 256, 256, 0, stream>>>(wr, wrb, CC_ * CC_);
  cvt_bf16_kernel<<<(CC_ * CC_ + 255) / 256, 256, 0, stream>>>(wl, wlb, CC_ * CC_);
  cvt_bf16_kernel<<<(KK_ * CC_ + 255) / 256, 256, 0, stream>>>(cw, cwb, KK_ * CC_);
  cvt_bf16_kernel<<<(CC_ * CC_ + 255) / 256, 256, 0, stream>>>(wfc, wfcb, CC_ * CC_);

  // 1) x NCHW fp32 -> NHWC bf16
  transpose_fwd_kernel<<<tg, tb, 0, stream>>>(x, xTb);

  // 2) conv1 (512->128) + BN + ReLU -> h1b
  gemm_bf16_kernel<<<dim3(NP_ / 64, C4_ / 64), 256, 0, stream>>>(
      xTb, CC_, w1b, CC_, nullptr, h1b, nullptr, C4_, NP_, CC_, 0, EPS_CB,
      g1, b1, m1, v1, nullptr, nullptr, nullptr);

  // 3) conv2 3x3 (128->128) + BN + ReLU -> h2b
  conv3x3_bn_relu_kernel<<<dim3(BB_ * HH_, C4_ / 64), 256, 0, stream>>>(
      h1b, w2b, h2b, g2, b2, m2, v2, EPS_CB);

  // 4) residual 1x1 (512->512) + BN -> x1T (fp32)
  gemm_bf16_kernel<<<dim3(NP_ / 64, CC_ / 64), 256, 0, stream>>>(
      xTb, CC_, wrb, CC_, x1T, nullptr, nullptr, CC_, NP_, CC_, 1, EPS_CB,
      gr, br, mr, vr, nullptr, nullptr, nullptr);

  // 5) conv3 (128->512) + BN + ReLU, accumulate into x1T; dual bf16 copy x1b
  gemm_bf16_kernel<<<dim3(NP_ / 64, CC_ / 64), 256, 0, stream>>>(
      h2b, C4_, w3b, C4_, x1T, x1b, nullptr, CC_, NP_, C4_, 2, EPS_CB,
      g3, b3, m3, v3, nullptr, nullptr, nullptr);

  // 6) LVC 1x1 (512->512) + BN + ReLU -> y (pixel-major) and yT (channel-major)
  gemm_bf16_kernel<<<dim3(NP_ / 64, CC_ / 64), 256, 0, stream>>>(
      x1b, CC_, wlb, CC_, nullptr, y, yT, CC_, NP_, CC_, 5, EPS_BN,
      gl, bl, ml, vl, nullptr, nullptr, nullptr);

  // 7) row |y|^2 and codeword |c|^2
  rowsq_bf16_kernel<<<NP_ / 8, 256, 0, stream>>>(y, xsq, NP_, CC_);
  rowsq_bf16_kernel<<<KK_ / 8, 256, 0, stream>>>(cwb, csq, KK_, CC_);

  // 8) dist = scale * (|y|^2 - 2 y.c + |c|^2)   (fp32, 65536x64)
  gemm_bf16_kernel<<<dim3(NP_ / 64, KK_ / 64), 256, 0, stream>>>(
      y, CC_, cwb, CC_, dist, nullptr, nullptr, KK_, NP_, CC_, 3, 0.f,
      nullptr, nullptr, nullptr, nullptr, xsq, csq, scl);

  // 9) softmax over codes -> code-major bf16 A^T
  softmax64_kernel<<<NP_ / 8, 256, 0, stream>>>(dist, AbT, NP_);

  // 10) asum[b,k] = sum_n A^T[b,k,n]
  asum_kernel<<<(BB_ * KK_) / 8, 256, 0, stream>>>(AbT, asum);

  // 11) enc aggregation + BN1d + ReLU -> encR
  enc_gemm_kernel<<<dim3(BB_, CC_ / 64), 256, 0, stream>>>(
      AbT, yT, asum, cw, ge, be_, me, ve, EPS_BN, encR);

  // 12) mean over codes -> en (bf16, padded to 64 rows)
  mean_codes_kernel<<<(BB_ * CC_ + 255) / 256, 256, 0, stream>>>(encR, en);

  // 13) fc gate: gamma = sigmoid(en @ wfc^T + bfc)
  gemm_bf16_kernel<<<dim3(1, CC_ / 64), 256, 0, stream>>>(
      en, CC_, wfcb, CC_, gam, nullptr, nullptr, CC_, BB_, CC_, 4, 0.f,
      nullptr, bfc, nullptr, nullptr, nullptr, nullptr, nullptr);

  // 14) out = relu(x1 * (1 + gamma)), NHWC -> NCHW
  final_gate_kernel<<<tg, tb, 0, stream>>>(x1T, gam, out);
}
